// DeformConv2d_84774064489045
// MI455X (gfx1250) — matrix operand order repaired and checked by reading, not compiled, not run on hardware
//
#include <hip/hip_runtime.h>

typedef __attribute__((ext_vector_type(16))) _Float16 v16h;
typedef __attribute__((ext_vector_type(8)))  float    v8f;
typedef __attribute__((ext_vector_type(4)))  unsigned u32x4;

namespace {
constexpr int CIN  = 128;
constexpr int COUT = 128;
constexpr int HW   = 96;          // H == W == HO == WO
constexpr int NSP  = HW * HW;     // 9216
constexpr int KTAPS = 9;
constexpr int BATCH = 8;
constexpr int BN = 128;           // n-tile per workgroup
constexpr int BK = 32;            // K-chunk (one WMMA depth)
constexpr int LDT = 40;           // padded LDS stride in halfs (80B, 16B aligned)
constexpr int NCHUNK = (CIN * KTAPS) / BK;   // 36

union Frag16 { v16h h; u32x4 q[2]; };

__device__ inline unsigned pack2(float a, float b) {
  union { _Float16 h[2]; unsigned u; } t;
  t.h[0] = (_Float16)a;
  t.h[1] = (_Float16)b;
  return t.u;
}
} // namespace

__launch_bounds__(256)
__global__ void deform_conv2d_wmma(const float* __restrict__ x,
                                   const float* __restrict__ off,
                                   const float* __restrict__ wgt,
                                   float* __restrict__ out) {
  // ping-pong LDS tiles: A = weights (co x kc), B = gathered samples (n x kc)
  __shared__ alignas(16) _Float16 As[2][COUT * LDT];
  __shared__ alignas(16) _Float16 Bs[2][BN * LDT];

  const int b    = blockIdx.y;
  const int n0   = blockIdx.x * BN;
  const int tid  = threadIdx.x;
  const int lane = tid & 31;
  const int wave = tid >> 5;          // M stripe: co = wave*16..+15

  // gather assignment: 2 threads per column, each covers 16 of 32 channels
  const int ncol  = tid & 127;
  const int chalf = tid >> 7;         // 0 or 1
  const int kcb   = chalf * 16;
  const int n  = n0 + ncol;
  const int ho = n / HW;
  const int wo = n - ho * HW;

  const float* xb = x   + (size_t)b * CIN * NSP;
  const float* ob = off + (size_t)b * 2 * KTAPS * NSP;

  v8f acc[8];
  #pragma unroll
  for (int i = 0; i < 8; ++i) acc[i] = {};

  // fragment-read geometry (contiguous k = 16*h + s, same order for A and B)
  const int l16   = lane & 15;
  const int koffA = (lane < 16) ? 0 : 16;  // A: K 0..15 vs 16..31
  const int koffB = (lane < 16) ? 0 : 16;  // B: K 0..15 vs 16..31
  const int arow  = (wave * 16 + l16) * LDT;

  // loop-carried bilinear geometry for the current tap k = t>>2
  float w00 = 0.f, w01 = 0.f, w10 = 0.f, w11 = 0.f;
  int   i00 = 0, i01 = 0, i10 = 0, i11 = 0;

  unsigned rA[8], rB[8];   // staged f16 pairs for next LDS store

  auto gather = [&](int t) {
    const int k  = t >> 2;
    const int c0 = (t & 3) * BK;
    if ((t & 3) == 0) {   // new tap: recompute bilinear geometry (uniform)
      const float py = (float)(ho - 1 + (k / 3)) + ob[(2 * k)     * NSP + n];
      const float px = (float)(wo - 1 + (k % 3)) + ob[(2 * k + 1) * NSP + n];
      const float y0f = floorf(py), x0f = floorf(px);
      const float ly = py - y0f, lx = px - x0f;
      const int y0 = (int)y0f, x0 = (int)x0f;
      const int y1 = y0 + 1,   x1 = x0 + 1;
      const float vy0 = (y0 >= 0 && y0 < HW) ? 1.f : 0.f;
      const float vy1 = (y1 >= 0 && y1 < HW) ? 1.f : 0.f;
      const float vx0 = (x0 >= 0 && x0 < HW) ? 1.f : 0.f;
      const float vx1 = (x1 >= 0 && x1 < HW) ? 1.f : 0.f;
      w00 = (1.f - ly) * (1.f - lx) * vy0 * vx0;
      w01 = (1.f - ly) * lx         * vy0 * vx1;
      w10 = ly         * (1.f - lx) * vy1 * vx0;
      w11 = ly         * lx         * vy1 * vx1;
      const int cy0 = min(max(y0, 0), HW - 1) * HW;
      const int cy1 = min(max(y1, 0), HW - 1) * HW;
      const int cx0 = min(max(x0, 0), HW - 1);
      const int cx1 = min(max(x1, 0), HW - 1);
      i00 = cy0 + cx0; i01 = cy0 + cx1;
      i10 = cy1 + cx0; i11 = cy1 + cx1;
    }
    // A: As[co][kc] = (f16) wgt[co][c0+kc][k]
    const float* wp = wgt + (size_t)ncol * (CIN * KTAPS) + (c0 + kcb) * KTAPS + k;
    #pragma unroll
    for (int j = 0; j < 8; ++j)
      rA[j] = pack2(wp[(2 * j) * KTAPS], wp[(2 * j + 1) * KTAPS]);
    // B: Bs[n][kc] = bilinear(x[b, c0+kc, py, px])
    const float* xc = xb + (size_t)(c0 + kcb) * NSP;
    #pragma unroll
    for (int j = 0; j < 8; ++j) {
      const float* p0 = xc + (size_t)(2 * j)     * NSP;
      const float* p1 = xc + (size_t)(2 * j + 1) * NSP;
      const float s0 = w00 * p0[i00] + w01 * p0[i01] + w10 * p0[i10] + w11 * p0[i11];
      const float s1 = w00 * p1[i00] + w01 * p1[i01] + w10 * p1[i10] + w11 * p1[i11];
      rB[j] = pack2(s0, s1);
    }
  };

  gather(0);   // prologue

  for (int t = 0; t < NCHUNK; ++t) {
    const int p = t & 1;

    // store staged tiles into ping-pong buffer p
    {
      u32x4 d0, d1;
      d0.x = rA[0]; d0.y = rA[1]; d0.z = rA[2]; d0.w = rA[3];
      d1.x = rA[4]; d1.y = rA[5]; d1.z = rA[6]; d1.w = rA[7];
      u32x4* da = (u32x4*)&As[p][ncol * LDT + kcb];
      da[0] = d0; da[1] = d1;
      u32x4 e0, e1;
      e0.x = rB[0]; e0.y = rB[1]; e0.z = rB[2]; e0.w = rB[3];
      e1.x = rB[4]; e1.y = rB[5]; e1.z = rB[6]; e1.w = rB[7];
      u32x4* db = (u32x4*)&Bs[p][ncol * LDT + kcb];
      db[0] = e0; db[1] = e1;
    }

    // issue next chunk's global gathers before the barrier: their latency
    // overlaps the barrier wait and this chunk's WMMAs (stores consume the
    // staging VGPRs at issue, so rewriting them is hazard-free)
    if (t + 1 < NCHUNK) gather(t + 1);

    __syncthreads();   // buffer p visible; prior reads of p (t-2) fenced by t-1

    // WMMA over this 32-deep K chunk
    Frag16 a;
    a.q[0] = *(const u32x4*)&As[p][arow + koffA];        // K = koffA..+7
    a.q[1] = *(const u32x4*)&As[p][arow + 8 + koffA];    // K = koffA+8..+15
    #pragma unroll
    for (int nt = 0; nt < 8; ++nt) {
      Frag16 bf;
      const int brow = (nt * 16 + l16) * LDT + koffB;
      bf.q[0] = *(const u32x4*)&Bs[p][brow];             // K = koffB..+7
      bf.q[1] = *(const u32x4*)&Bs[p][brow + 8];         // K = koffB+8..+15
      acc[nt] = __builtin_amdgcn_wmma_f32_16x16x32_f16(
          false, a.h, false, bf.h, (short)0, acc[nt], false, false);
    }
  }

  // epilogue: C/D layout -> lane L: VGPR v holds (M = v + (L>=16 ? 8:0), N = L%16)
  const int rbase = wave * 16 + ((lane >= 16) ? 8 : 0);
  #pragma unroll
  for (int nt = 0; nt < 8; ++nt) {
    const int col = n0 + nt * 16 + l16;
    float* op = out + ((size_t)b * COUT + rbase) * NSP + col;
    #pragma unroll
    for (int v = 0; v < 8; ++v)
      op[(size_t)v * NSP] = acc[nt][v];
  }
}

extern "C" void kernel_launch(void* const* d_in, const int* in_sizes, int n_in,
                              void* d_out, int out_size, void* d_ws, size_t ws_size,
                              hipStream_t stream) {
  const float* x   = (const float*)d_in[0];   // [8,128,96,96]
  const float* off = (const float*)d_in[1];   // [8,18,96,96]
  const float* w   = (const float*)d_in[2];   // [128,128,3,3]
  float* out = (float*)d_out;                 // [8,128,96,96]
  (void)in_sizes; (void)n_in; (void)out_size; (void)d_ws; (void)ws_size;

  dim3 grid(NSP / BN, BATCH);   // 72 x 8
  deform_conv2d_wmma<<<grid, 256, 0, stream>>>(x, off, w, out);
}
